// BiLevelRoutingAttention_37391985279424
// MI455X (gfx1250) — matrix operand order repaired and checked by reading, not compiled, not run on hardware
//
#include <hip/hip_runtime.h>

typedef __attribute__((ext_vector_type(16))) _Float16 v16h;
typedef __attribute__((ext_vector_type(8)))  _Float16 v8h;
typedef __attribute__((ext_vector_type(8)))  float    v8f;
typedef __attribute__((ext_vector_type(4)))  float    v4f;

// Problem constants (from reference)
#define TT 4
#define BB 4
#define LL 4096
#define CC 256
#define C3 768
#define NWIN 8
#define WIN 512
#define HEADS 4
#define DH 64
#define TOPK 4
#define MROWS 65536            // T*B*L
#define SPIKE_TH 2.0f          // spike = H(x/tau - v_th) = (x >= 2)

// Build a 16-half WMMA fragment from two contiguous 8-half (128-bit) runs.
// Matches the wave32 f16 A-layout (runs at K=half*8 and K=16+half*8) and the
// B-layout when B is stored transposed (same runs at K=half*8 and K=16+half*8).
__device__ __forceinline__ v16h frag2(const _Float16* __restrict__ p0,
                                      const _Float16* __restrict__ p1) {
    v8h r0 = *(const v8h*)p0;
    v8h r1 = *(const v8h*)p1;
    v16h a;
#pragma unroll
    for (int e = 0; e < 8; ++e) { a[e] = r0[e]; a[8 + e] = r1[e]; }
    return a;
}

// =====================================================================
// Pre-pass: x fp32 -> xh fp16 (row-major [65536][256])
// =====================================================================
__global__ void cvt_x_kernel(const float* __restrict__ x, _Float16* __restrict__ xh) {
    const size_t i = ((size_t)blockIdx.x * blockDim.x + threadIdx.x) * 8;
    v4f a = *(const v4f*)(x + i);
    v4f b = *(const v4f*)(x + i + 4);
    v8h o;
#pragma unroll
    for (int e = 0; e < 4; ++e) { o[e] = (_Float16)a[e]; o[4 + e] = (_Float16)b[e]; }
    *(v8h*)(xh + i) = o;
}

// Pre-pass: W [K][N] fp32 -> WT [N][K] fp16
__global__ void cvt_wT_kernel(const float* __restrict__ W, _Float16* __restrict__ WT,
                              int K, int N) {
    const int i = blockIdx.x * blockDim.x + threadIdx.x;   // over K*N, coalesced read
    const int k = i / N, n = i - k * N;
    WT[(size_t)n * K + k] = (_Float16)W[i];
}

// =====================================================================
// Kernel 1: region[b,n,c] = sum_t sum_w x[t,b,n*512+w,c]   (fp32 exact path)
// =====================================================================
__global__ void region_kernel(const float* __restrict__ x, float* __restrict__ region) {
    const int bn = blockIdx.x;          // 0..31
    const int b = bn >> 3, n = bn & 7;
    const int c = threadIdx.x;          // 0..255
    float s = 0.0f;
    for (int t = 0; t < TT; ++t) {
        const float* p = x + (((size_t)(t * BB + b) * LL + n * WIN) * CC) + c;
        for (int w = 0; w < WIN; ++w) s += p[(size_t)w * CC];
    }
    region[bn * CC + c] = s;
}

// =====================================================================
// Kernel 2: attn_r = region @ region^T per batch; top-4 indices per row
// =====================================================================
__global__ void route_kernel(const float* __restrict__ region, int* __restrict__ idx) {
    __shared__ float s_reg[NWIN * CC];
    __shared__ float s_attn[NWIN][NWIN];
    const int b = blockIdx.x;
    for (int i = threadIdx.x; i < NWIN * CC; i += blockDim.x)
        s_reg[i] = region[b * NWIN * CC + i];
    __syncthreads();
    if (threadIdx.x < 64) {
        const int n = threadIdx.x >> 3, m = threadIdx.x & 7;
        float d = 0.0f;
        for (int c = 0; c < CC; ++c) d += s_reg[n * CC + c] * s_reg[m * CC + c];
        s_attn[n][m] = d;
    }
    __syncthreads();
    if (threadIdx.x < NWIN) {
        const int n = threadIdx.x;
        float v[NWIN];
        for (int m = 0; m < NWIN; ++m) v[m] = s_attn[n][m];
        for (int kk = 0; kk < TOPK; ++kk) {
            int best = 0; float bv = v[0];
            for (int m = 1; m < NWIN; ++m)
                if (v[m] > bv) { bv = v[m]; best = m; }  // stable: lowest index on ties
            idx[(b * NWIN + n) * TOPK + kk] = best;
            v[best] = -3.4e38f;
        }
    }
}

// =====================================================================
// Kernel 3: qkv = xh @ WqT^T + b ; LIF spike.
//   q -> row-major [row][c];  k,v -> transposed [c][row] for the kv einsum.
// one wave per 16x16 tile: M tiles 4096 x N tiles 48; K=256 fully unrolled
// =====================================================================
__global__ void qkv_kernel(const _Float16* __restrict__ xh, const _Float16* __restrict__ WqT,
                           const float* __restrict__ bq,
                           _Float16* __restrict__ qh, _Float16* __restrict__ kT,
                           _Float16* __restrict__ vT) {
    const int gw = blockIdx.x * 8 + (threadIdx.x >> 5);  // 196608 wave-tiles
    const int nt = gw % 48, mt = gw / 48;
    const int ln = threadIdx.x & 31, half = ln >> 4, l15 = ln & 15;
    const _Float16* arow = xh + (size_t)(mt * 16 + l15) * CC;
    const int gcol = nt * 16 + l15;
    const _Float16* brow = WqT + (size_t)gcol * CC;
    v8f acc = {};
#pragma unroll
    for (int kb = 0; kb < CC; kb += 32) {
        v16h a  = frag2(arow + kb + half * 8,  arow + kb + 16 + half * 8);
        v16h bm = frag2(brow + kb + half * 8,  brow + kb + 16 + half * 8);
        acc = __builtin_amdgcn_wmma_f32_16x16x32_f16(false, a, false, bm,
                                                     (short)0, acc, false, false);
    }
    const float bias = bq[gcol];
#pragma unroll
    for (int r = 0; r < 8; ++r) {
        const int row = mt * 16 + r + 8 * half;
        const _Float16 s = (acc[r] + bias >= SPIKE_TH) ? (_Float16)1.0f : (_Float16)0.0f;
        if (gcol < CC)          qh[(size_t)row * CC + gcol] = s;
        else if (gcol < 2 * CC) kT[(size_t)(gcol - CC) * MROWS + row] = s;
        else                    vT[(size_t)(gcol - 2 * CC) * MROWS + row] = s;
    }
}

// =====================================================================
// Kernel 4: kv[d,e] = sum_s k_g[s,d]*v_g[s,e] over 4 routed windows (s=2048)
// kT/vT are [c][row] so both fragments are contiguous 128-bit runs along s.
// one block (16 waves) per (t,b,n,h); each wave one 16x16 tile of 64x64.
// Output stored TRANSPOSED: kvT[e][d] so kernel 5's B-fragments are contiguous.
// =====================================================================
__global__ void kv_kernel(const _Float16* __restrict__ kT, const _Float16* __restrict__ vT,
                          const int* __restrict__ idx, _Float16* __restrict__ kvT) {
    const int id = blockIdx.x;                       // 512 = T*B*N*H
    const int h = id & 3, n = (id >> 2) & 7, b = (id >> 5) & 3, t = id >> 7;
    const int ln = threadIdx.x & 31, half = ln >> 4, l15 = ln & 15;
    const int wv = threadIdx.x >> 5;                 // 0..15
    const int dt = (wv >> 2) * 16, et = (wv & 3) * 16;
    int wb[TOPK];
#pragma unroll
    for (int j = 0; j < TOPK; ++j)
        wb[j] = (t * BB + b) * LL + idx[(b * NWIN + n) * TOPK + j] * WIN;
    const _Float16* acol = kT + (size_t)(h * DH + dt + l15) * MROWS;  // A row (d) = lane M
    const _Float16* bcol = vT + (size_t)(h * DH + et + l15) * MROWS;  // B col (e) = lane N
    v8f acc = {};
#pragma unroll 4
    for (int step = 0; step < 64; ++step) {          // K = 2048 in steps of 32
        const int sb = step * 32;
        const int base = wb[sb >> 9] + (sb & (WIN - 1));  // runs never cross a window
        if (step + 1 < 64) {                               // gfx1250 global_prefetch path
            const int nb = (step + 1) * 32;
            const int pbase = wb[nb >> 9] + (nb & (WIN - 1));
            __builtin_prefetch(acol + pbase, 0, 1);
            __builtin_prefetch(bcol + pbase, 0, 1);
        }
        v16h a  = frag2(acol + base + half * 8,  acol + base + 16 + half * 8);
        v16h bm = frag2(bcol + base + half * 8,  bcol + base + 16 + half * 8);
        acc = __builtin_amdgcn_wmma_f32_16x16x32_f16(false, a, false, bm,
                                                     (short)0, acc, false, false);
    }
    _Float16* dst = kvT + (size_t)id * (DH * DH);
#pragma unroll
    for (int r = 0; r < 8; ++r)                      // counts <= 2048: exact in fp16
        dst[(et + l15) * DH + dt + r + 8 * half] = (_Float16)acc[r];
}

// =====================================================================
// Kernel 5: out = q @ kv per (t,b,n,h): [512x64]@[64x64] -> attn fp16 row-major
// =====================================================================
__global__ void out_kernel(const _Float16* __restrict__ qh, const _Float16* __restrict__ kvT,
                           _Float16* __restrict__ attn) {
    const int gw = blockIdx.x * 8 + (threadIdx.x >> 5);  // 65536 wave-tiles
    const int tile = gw & 127, id = gw >> 7;
    const int h = id & 3, n = (id >> 2) & 7, b = (id >> 5) & 3, t = id >> 7;
    const int et = (tile & 3) * 16, wt = (tile >> 2) * 16;
    const int ln = threadIdx.x & 31, half = ln >> 4, l15 = ln & 15;
    const size_t rowbase = (size_t)(t * BB + b) * LL + n * WIN + wt;
    const _Float16* arow = qh + (rowbase + l15) * CC + h * DH;
    const _Float16* brow = kvT + (size_t)id * (DH * DH) + (et + l15) * DH;
    v8f acc = {};
#pragma unroll
    for (int kb = 0; kb < DH; kb += 32) {            // K = 64
        v16h a  = frag2(arow + kb + half * 8,  arow + kb + 16 + half * 8);
        v16h bm = frag2(brow + kb + half * 8,  brow + kb + 16 + half * 8);
        acc = __builtin_amdgcn_wmma_f32_16x16x32_f16(false, a, false, bm,
                                                     (short)0, acc, false, false);
    }
#pragma unroll
    for (int r = 0; r < 8; ++r)
        attn[(rowbase + r + 8 * half) * CC + h * DH + et + l15] = (_Float16)acc[r];
}

// =====================================================================
// Kernel 6: lif(attn @ W_proj + b_proj) -> fp32 {0,1} output
// =====================================================================
__global__ void proj_kernel(const _Float16* __restrict__ attn, const _Float16* __restrict__ WpT,
                            const float* __restrict__ bp, float* __restrict__ out) {
    const int gw = blockIdx.x * 8 + (threadIdx.x >> 5);  // 65536 wave-tiles
    const int nt = gw & 15, mt = gw >> 4;
    const int ln = threadIdx.x & 31, half = ln >> 4, l15 = ln & 15;
    const _Float16* arow = attn + (size_t)(mt * 16 + l15) * CC;
    const _Float16* brow = WpT + (size_t)(nt * 16 + l15) * CC;
    v8f acc = {};
#pragma unroll
    for (int kb = 0; kb < CC; kb += 32) {
        v16h a  = frag2(arow + kb + half * 8,  arow + kb + 16 + half * 8);
        v16h bm = frag2(brow + kb + half * 8,  brow + kb + 16 + half * 8);
        acc = __builtin_amdgcn_wmma_f32_16x16x32_f16(false, a, false, bm,
                                                     (short)0, acc, false, false);
    }
    const float bias = bp[nt * 16 + l15];
#pragma unroll
    for (int r = 0; r < 8; ++r)
        out[(size_t)(mt * 16 + r + 8 * half) * CC + nt * 16 + l15] =
            (acc[r] + bias >= SPIKE_TH) ? 1.0f : 0.0f;
}

// =====================================================================
extern "C" void kernel_launch(void* const* d_in, const int* in_sizes, int n_in,
                              void* d_out, int out_size, void* d_ws, size_t ws_size,
                              hipStream_t stream) {
    (void)in_sizes; (void)n_in; (void)out_size; (void)ws_size;
    const float* x     = (const float*)d_in[0];
    const float* W_qkv = (const float*)d_in[1];
    const float* b_qkv = (const float*)d_in[2];
    const float* W_proj= (const float*)d_in[3];
    const float* b_proj= (const float*)d_in[4];
    float* out = (float*)d_out;

    char* ws = (char*)d_ws;
    const size_t M = (size_t)MROWS;
    size_t off = 0;
    float*    region = (float*)(ws + off);    off += (size_t)BB * NWIN * CC * 4;
    int*      idx    = (int*)(ws + off);      off += (size_t)BB * NWIN * TOPK * 4;
    off = (off + 255) & ~(size_t)255;
    _Float16* xh   = (_Float16*)(ws + off); off += M * CC * 2;            // 32 MB
    _Float16* WqT  = (_Float16*)(ws + off); off += (size_t)C3 * CC * 2;   // 384 KB
    _Float16* WpT  = (_Float16*)(ws + off); off += (size_t)CC * CC * 2;   // 128 KB
    off = (off + 255) & ~(size_t)255;
    _Float16* qh   = (_Float16*)(ws + off); off += M * CC * 2;            // 32 MB
    _Float16* kT   = (_Float16*)(ws + off); off += M * CC * 2;
    _Float16* vT   = (_Float16*)(ws + off); off += M * CC * 2;
    _Float16* kvT  = (_Float16*)(ws + off); off += (size_t)512 * DH * DH * 2;
    _Float16* attn = (_Float16*)(ws + off); off += M * CC * 2;

    cvt_x_kernel<<<(int)(M * CC / 8 / 256), 256, 0, stream>>>(x, xh);
    cvt_wT_kernel<<<(CC * C3) / 256, 256, 0, stream>>>(W_qkv, WqT, CC, C3);
    cvt_wT_kernel<<<(CC * CC) / 256, 256, 0, stream>>>(W_proj, WpT, CC, CC);
    region_kernel<<<BB * NWIN, CC, 0, stream>>>(x, region);
    route_kernel<<<BB, 64, 0, stream>>>(region, idx);
    qkv_kernel<<<(4096 * 48) / 8, 256, 0, stream>>>(xh, WqT, b_qkv, qh, kT, vT);
    kv_kernel<<<512, 512, 0, stream>>>(kT, vT, idx, kvT);
    out_kernel<<<65536 / 8, 256, 0, stream>>>(qh, kvT, attn);
    proj_kernel<<<65536 / 8, 256, 0, stream>>>(attn, W_proj ? WpT : WpT, b_proj, out);
}
